// SAGE_25125558682200
// MI455X (gfx1250) — matrix operand order repaired and checked by reading, not compiled, not run on hardware
//
#include <hip/hip_runtime.h>

// ---------------------------------------------------------------------------
// GraphSAGE 2-layer forward for MI455X (gfx1250, wave32, WMMA bf16)
//   layer: h = act( [segment_mean(gather)] @ Wl^T + b + x_tgt @ Wr^T )
//   fused as one K=512 GEMM against Bpack = [Wl ; Wr] in bf16 WMMA fragments
//   64-row M tiles per block: B fragment loaded once per k-step, reused for
//   4 M-subtiles (4 independent accumulator chains). A tile is staged into
//   LDS in *fragment order* so the inner loop uses one base VGPR + immediate
//   DS offsets only (no per-step address VALU, no WMMA->VALU hazard NOPs),
//   and each ds_load_b128 is bank-conflict-free.
// ---------------------------------------------------------------------------

#define N1C 120000
#define N2C 12000
#define INC 256
#define KTOT 512          // 2 * 256 (agg half + target half)
#define KSTEPS 16         // KTOT / 32
#define MROWS 64          // rows per block
#define MT 4              // 16-row M-subtiles per wave

typedef __attribute__((ext_vector_type(16))) __bf16 v16bf;
typedef __attribute__((ext_vector_type(8)))  float  v8f;

union ABFrag {
    int4  i4[2];   // 2 x 16B = 32B
    v16bf v;       // 16 bf16
};

// round-to-nearest-even f32 -> bf16 bits
__device__ __forceinline__ unsigned short f2bf(float f) {
    unsigned int u = __float_as_uint(f);
    u = (u + 0x7FFFu + ((u >> 16) & 1u)) >> 16;
    return (unsigned short)u;
}

// ---------------------------------------------------------------------------
// Pack B = [Wl ; Wr] (K=512 x N) into per-lane WMMA bf16 B fragments.
// 16-bit B 32x16 layout: lane L holds column n=L%16; lane group g=L/16:
//   dword d (0..7) holds K = ks*32 + 16*g + 2*d , +1   (contiguous k order used here; A in LDS matches)
// dword index in bpack == flattened thread id t = ((nt*16+ks)*32+lane)*8+d
// ---------------------------------------------------------------------------
__global__ void sage_pack_w(const float* __restrict__ Wl,
                            const float* __restrict__ Wr,
                            unsigned int* __restrict__ bpack, int N) {
    int t = blockIdx.x * blockDim.x + threadIdx.x;
    int total = (N / 16) * KSTEPS * 32 * 8;
    if (t >= total) return;
    int d    = t & 7;
    int lane = (t >> 3) & 31;
    int ks   = (t >> 8) & 15;
    int nt   = t >> 12;
    int n    = nt * 16 + (lane & 15);
    int g    = lane >> 4;
    int k    = ks * 32 + 16 * g + 2 * d;            // even; k,k+1 same half
    const float* Wsel = (k < INC) ? Wl : Wr;
    int kk = (k < INC) ? k : (k - INC);
    float f0 = Wsel[(long long)n * INC + kk];
    float f1 = Wsel[(long long)n * INC + kk + 1];
    bpack[t] = (unsigned int)f2bf(f0) | ((unsigned int)f2bf(f1) << 16);
}

// ---------------------------------------------------------------------------
// Edge scatter: one thread per (edge, 4-channel chunk); f32 global atomics.
// ---------------------------------------------------------------------------
__global__ void sage_scatter(const float* __restrict__ x,
                             const long long* __restrict__ src,
                             const long long* __restrict__ dst,
                             float* __restrict__ agg,
                             float* __restrict__ cnt, int E) {
    long long t = (long long)blockIdx.x * blockDim.x + threadIdx.x;
    long long e = t >> 6;
    if (e >= E) return;
    int c4 = (int)(t & 63) << 2;
    long long s = src[e];
    long long d = dst[e];
    float4 v = *(const float4*)(x + s * INC + c4);
    float* ap = agg + d * INC + c4;
    atomicAdd(ap + 0, v.x);
    atomicAdd(ap + 1, v.y);
    atomicAdd(ap + 2, v.z);
    atomicAdd(ap + 3, v.w);
    if (c4 == 0) atomicAdd(cnt + d, 1.0f);
}

// ---------------------------------------------------------------------------
// Fused SAGE GEMM: out[m, 0:N] = [agg[m]/max(cnt,1) | xt[m]] @ Bpack + bias
// Block: one 64-row M tile; N/16 waves (1 wave per 16-col N tile), each wave
// computes 4 M-subtiles sharing each B fragment (4 acc chains for ILP).
//
// LDS A layout (fragment order, 64 KB total): per (mt, ks) a 1024 B group at
//   group = (mt*KSTEPS + ks) * 1024 B
//   lane L half-0 (K = ks*32 + 16g .. +7   of row mt*16 + (L&15)) at +L*16
//   lane L half-1 (K = ks*32 + 16g + 8 ..)                       at +512+L*16
// -> inner-loop DS addresses are (lane*16) + immediate, max imm 65,024 B.
// -> per 16B load, half-wave lanes land on dwords {0,4,...,60} mod 64: no
//    bank conflicts.
// ---------------------------------------------------------------------------
__global__ __launch_bounds__(512) void sage_gemm(
    const float* __restrict__ agg, const float* __restrict__ cnt,
    const float* __restrict__ xt, const unsigned short* __restrict__ bpack,
    const float* __restrict__ bias, float* __restrict__ out,
    int M, int N, int relu) {
    __shared__ unsigned short sA[MROWS * KTOT];   // 64 KB, fragment-ordered

    const int tid   = threadIdx.x;
    const int mbase = blockIdx.x * MROWS;

    // ---- stage A tile (f32 -> bf16, agg half scaled by 1/cnt), writing
    //      each 16B chunk straight into its fragment slot ----
    for (int chunk = tid; chunk < MROWS * KTOT / 8; chunk += blockDim.x) {
        int rr  = chunk >> 6;                  // row in tile (0..63)
        int c0  = (chunk & 63) << 3;           // K base of this 8-elt chunk
        int row = mbase + rr;
        int4 st;
        if (row < M) {
            float4 v0, v1;
            float scale = 1.0f;
            if (c0 < INC) {
                scale = 1.0f / fmaxf(cnt[row], 1.0f);
                const float4* p = (const float4*)(agg + (long long)row * INC + c0);
                v0 = p[0]; v1 = p[1];
            } else {
                const float4* p = (const float4*)(xt + (long long)row * INC + (c0 - INC));
                v0 = p[0]; v1 = p[1];
            }
            st.x = (int)((unsigned int)f2bf(v0.x * scale) | ((unsigned int)f2bf(v0.y * scale) << 16));
            st.y = (int)((unsigned int)f2bf(v0.z * scale) | ((unsigned int)f2bf(v0.w * scale) << 16));
            st.z = (int)((unsigned int)f2bf(v1.x * scale) | ((unsigned int)f2bf(v1.y * scale) << 16));
            st.w = (int)((unsigned int)f2bf(v1.z * scale) | ((unsigned int)f2bf(v1.w * scale) << 16));
        } else {
            st.x = 0; st.y = 0; st.z = 0; st.w = 0;   // pad rows -> zeros
        }
        // (row, c0) -> fragment slot
        int ks   = c0 >> 5;                    // k-step
        int rem  = c0 & 31;                    // 0,8,16,24
        int g    = rem >> 4;                   // lane group
        int half = (rem >> 3) & 1;             // 0: K+16g+0..7, 1: K+16g+8..15
        int mt   = rr >> 4;
        int r    = rr & 15;
        int lane = (g << 4) | r;
        int off  = ((mt * KSTEPS + ks) << 9) + (half << 8) + (lane << 3); // ushorts
        *(int4*)(sA + off) = st;
    }
    __syncthreads();

    // ---- WMMA main loop: wave wv owns N-tile wv; 4 M-subtiles per wave ----
    const int wv   = tid >> 5;
    const int lane = tid & 31;
    const int r    = lane & 15;
    const int g    = lane >> 4;

    v8f acc[MT] = {{}, {}, {}, {}};
    const unsigned short* bp = bpack + (long long)wv * KSTEPS * 32 * 16;
    const unsigned short* abase = sA + (lane << 3);   // lane*16 bytes

#pragma unroll
    for (int ks = 0; ks < KSTEPS; ++ks) {
        ABFrag b;
        const int4* bsrc = (const int4*)(bp + (ks * 32 + lane) * 16);
        b.i4[0] = bsrc[0];
        b.i4[1] = bsrc[1];

        ABFrag a[MT];
#pragma unroll
        for (int mt = 0; mt < MT; ++mt) {      // immediate-offset DS loads
            const int goff = (mt * KSTEPS + ks) << 9;     // ushorts
            a[mt].i4[0] = *(const int4*)(abase + goff);          // half 0
            a[mt].i4[1] = *(const int4*)(abase + goff + 256);    // half 1
        }
#pragma unroll
        for (int mt = 0; mt < MT; ++mt) {
            acc[mt] = __builtin_amdgcn_wmma_f32_16x16x32_bf16(
                false, a[mt].v, false, b.v, (short)0, acc[mt], false, false);
        }
    }

    // ---- epilogue: C/D layout lane n = wv*16 + (lane&15), M = g*8 + v ----
    const int n  = wv * 16 + r;
    const float bn = bias[n];
#pragma unroll
    for (int mt = 0; mt < MT; ++mt) {
        const int m0 = mbase + mt * 16 + (g << 3);
#pragma unroll
        for (int v = 0; v < 8; ++v) {
            int m = m0 + v;
            if (m < M) {
                float val = acc[mt][v] + bn;
                if (relu) val = fmaxf(val, 0.0f);
                out[(long long)m * N + n] = val;
            }
        }
    }
}

// ---------------------------------------------------------------------------
// Row log-softmax over 64 cols: one wave32 per row, shfl_xor reductions.
// ---------------------------------------------------------------------------
__global__ void sage_logsoftmax(float* __restrict__ out, int M) {
    int row = blockIdx.x * (blockDim.x >> 5) + (threadIdx.x >> 5);
    if (row >= M) return;
    int lane = threadIdx.x & 31;
    float* p = out + (long long)row * 64;
    float a = p[lane];
    float b = p[lane + 32];
    float mx = fmaxf(a, b);
#pragma unroll
    for (int off = 16; off > 0; off >>= 1)
        mx = fmaxf(mx, __shfl_xor(mx, off, 32));
    float s = expf(a - mx) + expf(b - mx);
#pragma unroll
    for (int off = 16; off > 0; off >>= 1)
        s += __shfl_xor(s, off, 32);
    float l = mx + logf(s);
    p[lane]      = a - l;
    p[lane + 32] = b - l;
}

// ---------------------------------------------------------------------------
extern "C" void kernel_launch(void* const* d_in, const int* in_sizes, int n_in,
                              void* d_out, int out_size, void* d_ws, size_t ws_size,
                              hipStream_t stream) {
    const float*      x    = (const float*)d_in[0];
    const long long*  src0 = (const long long*)d_in[1];
    const long long*  dst0 = (const long long*)d_in[2];
    const long long*  src1 = (const long long*)d_in[3];
    const long long*  dst1 = (const long long*)d_in[4];
    const float*      Wl0  = (const float*)d_in[5];
    const float*      b0   = (const float*)d_in[6];
    const float*      Wr0  = (const float*)d_in[7];
    const float*      Wl1  = (const float*)d_in[8];
    const float*      b1   = (const float*)d_in[9];
    const float*      Wr1  = (const float*)d_in[10];
    const int E0 = in_sizes[1];
    const int E1 = in_sizes[3];

    // workspace layout (all 16B-aligned); agg0 is reused as h after GEMM0
    char* w = (char*)d_ws;
    size_t off = 0;
    float* agg0 = (float*)(w + off); off += (size_t)N1C * INC * 4;  // also h
    float* cnt0 = (float*)(w + off); off += (size_t)N1C * 4;
    float* agg1 = (float*)(w + off); off += (size_t)N2C * INC * 4;
    float* cnt1 = (float*)(w + off); off += (size_t)N2C * 4;
    unsigned short* b0p = (unsigned short*)(w + off); off += (size_t)KTOT * 256 * 2;
    unsigned short* b1p = (unsigned short*)(w + off); off += (size_t)KTOT * 64 * 2;
    if (ws_size < off) return;  // deterministic guard

    // zero accumulators (agg0+cnt0 contiguous, agg1+cnt1 contiguous)
    hipMemsetAsync(agg0, 0, (size_t)N1C * INC * 4 + (size_t)N1C * 4, stream);
    hipMemsetAsync(agg1, 0, (size_t)N2C * INC * 4 + (size_t)N2C * 4, stream);

    // pack weights into WMMA B fragments (bf16)
    sage_pack_w<<<256, 256, 0, stream>>>(Wl0, Wr0, (unsigned int*)b0p, 256);
    sage_pack_w<<<64, 256, 0, stream>>>(Wl1, Wr1, (unsigned int*)b1p, 64);

    // layer 0: scatter-mean + fused GEMM + ReLU  (h aliases agg0)
    long long tw0 = (long long)E0 * 64;
    sage_scatter<<<(unsigned)((tw0 + 255) / 256), 256, 0, stream>>>(
        x, src0, dst0, agg0, cnt0, E0);
    sage_gemm<<<(N1C + MROWS - 1) / MROWS, 512, 0, stream>>>(
        agg0, cnt0, /*xt=*/x, b0p, b0, /*out=h=*/agg0, N1C, 256, /*relu=*/1);

    // layer 1: scatter-mean + fused GEMM (pre-softmax straight into d_out)
    long long tw1 = (long long)E1 * 64;
    sage_scatter<<<(unsigned)((tw1 + 255) / 256), 256, 0, stream>>>(
        /*x=h=*/agg0, src1, dst1, agg1, cnt1, E1);
    sage_gemm<<<(N2C + MROWS - 1) / MROWS, 128, 0, stream>>>(
        agg1, cnt1, /*xt=h=*/agg0, b1p, b1, (float*)d_out, N2C, 64, /*relu=*/0);

    // row-wise log-softmax in place on d_out
    sage_logsoftmax<<<(N2C + 7) / 8, 256, 0, stream>>>((float*)d_out, N2C);
}
